// MambaFallback_52037823758428
// MI455X (gfx1250) — matrix operand order repaired and checked by reading, not compiled, not run on hardware
//
#include <hip/hip_runtime.h>

typedef __bf16 bf16_t;
typedef __attribute__((ext_vector_type(16))) __bf16 v16bf;
typedef __attribute__((ext_vector_type(8)))  __bf16 v8bf;
typedef __attribute__((ext_vector_type(8)))  float  v8f;

#define D_MODEL 1024
#define TWO_D   2048
#define SEQ     4096
#define BATCH   4
#define M_TOTAL (BATCH * SEQ)   // 16384 rows

// Block tile: 128(M) x 64(N), K-step 32.  8 waves: 4 along M x 2 along N,
// each wave owns 2x2 WMMA 16x16 subtiles per B-matrix.
#define BM 128
#define BN 64
#define BK 32
#define LDA 40   // LDS pitch (bf16): 80 B -> 16B-aligned b128 chunks, bank spread
#define LDB 40
#define NK  (D_MODEL / BK)      // 32 k-steps

// ---- CDNA5 async global->LDS copy (ASYNCcnt-tracked), per-lane 16 B ----
#define ASYNC_LOAD_B128(ldsoff, gptr)                                   \
    asm volatile("global_load_async_to_lds_b128 %0, %1, off"            \
                 :: "v"(ldsoff), "v"(gptr) : "memory")
#define WAIT_ASYNC(imm)                                                 \
    asm volatile("s_wait_asynccnt " #imm ::: "memory")

__device__ __forceinline__ unsigned lds_off(const void* p) {
    // Flat shared-aperture addresses carry the LDS byte offset in the low 32 bits.
    return (unsigned)(unsigned long long)p;
}

__device__ __forceinline__ v8f vzero8() {
    v8f z;
#pragma unroll
    for (int i = 0; i < 8; ++i) z[i] = 0.0f;
    return z;
}

// A-matrix 16x32 bf16 fragment (ISA 7.12.2): lanes 0-15 hold K=[0..7],[16..23];
// lanes 16-31 hold K=[8..15],[24..31].
__device__ __forceinline__ v16bf frag_a(const bf16_t* row, int half) {
    v8bf lo = *(const v8bf*)(row + half * 8);
    v8bf hi = *(const v8bf*)(row + 16 + half * 8);
    v16bf r;
#pragma unroll
    for (int i = 0; i < 8; ++i) { r[i] = lo[i]; r[i + 8] = hi[i]; }
    return r;
}

// B-matrix 32x16 bf16 fragment: lanes 0-15 hold K=[0..7],[16..23]; lanes 16-31 K=[8..15],[24..31].
// LDS tile stored transposed ([n][k]) -> two contiguous b128 loads.
__device__ __forceinline__ v16bf frag_b(const bf16_t* col, int half) {
    v8bf lo = *(const v8bf*)(col + half * 8);
    v8bf hi = *(const v8bf*)(col + 16 + half * 8);
    v16bf r;
#pragma unroll
    for (int i = 0; i < 8; ++i) { r[i] = lo[i]; r[i + 8] = hi[i]; }
    return r;
}

// ---------------------------------------------------------------------------
// One-time prep kernels: fp32 -> bf16 (and transpose for weights so GEMM
// staging is a contiguous async copy).
// ---------------------------------------------------------------------------
__global__ __launch_bounds__(256)
void cvt_f32_to_bf16(const float* __restrict__ src, bf16_t* __restrict__ dst) {
    int i = (blockIdx.x * 256 + threadIdx.x) * 8;
    float4 a = *(const float4*)(src + i);
    float4 b = *(const float4*)(src + i + 4);
    bf16_t t[8];
    t[0] = (bf16_t)a.x; t[1] = (bf16_t)a.y; t[2] = (bf16_t)a.z; t[3] = (bf16_t)a.w;
    t[4] = (bf16_t)b.x; t[5] = (bf16_t)b.y; t[6] = (bf16_t)b.z; t[7] = (bf16_t)b.w;
    *(uint4*)(dst + i) = *(const uint4*)t;
}

// W[K][N] fp32 -> WT[N][K] bf16, 32x32 tiles via LDS.
__global__ __launch_bounds__(256)
void transpose_cvt(const float* __restrict__ W, bf16_t* __restrict__ WT,
                   int K, int N) {
    __shared__ float tile[32][33];
    const int nb = blockIdx.x * 32;
    const int kb = blockIdx.y * 32;
    const int tx = threadIdx.x & 31;
    const int ty = threadIdx.x >> 5;        // 0..7
#pragma unroll
    for (int r = 0; r < 32; r += 8)
        tile[ty + r][tx] = W[(size_t)(kb + ty + r) * N + nb + tx];
    __syncthreads();
#pragma unroll
    for (int r = 0; r < 32; r += 8)
        WT[(size_t)(nb + ty + r) * K + kb + tx] = (bf16_t)tile[tx][ty + r];
}

// ---------------------------------------------------------------------------
// Kernel 1: gated projection. G = sigmoid(x@Wg + bg) * silu(x@Wh + bh)
// Double-buffered LDS fed by async global->LDS copies.
// ---------------------------------------------------------------------------
__global__ __launch_bounds__(256)
void mamba_gemm1_gated(const bf16_t* __restrict__ Xb,    // [M,1024] bf16
                       const bf16_t* __restrict__ WinT,  // [2048,1024] bf16 (n-major)
                       const float* __restrict__ bin,    // [2048]
                       float* __restrict__ G)            // [M,1024] fp32
{
    __shared__ __align__(16) bf16_t lA [2][BM * LDA];
    __shared__ __align__(16) bf16_t lBg[2][BN * LDB];
    __shared__ __align__(16) bf16_t lBh[2][BN * LDB];

    const int tid  = threadIdx.x;
    const int m0   = blockIdx.x * BM;
    const int n0   = blockIdx.y * BN;
    const int lane = tid & 31;
    const int w    = tid >> 5;
    const int wm   = (w & 3) * 32;
    const int wn   = (w >> 2) * 32;
    const int half = lane >> 4;
    const int id16 = lane & 15;

    // Per-thread staging coordinates (4 async b128 ops per wave per stage).
    const int arow0 = tid >> 2,        ac0 = tid & 3;          // A part 0
    const int arow1 = (tid + 256) >> 2, ac1 = (tid + 256) & 3; // A part 1
    const int brow  = tid >> 2,        bc  = tid & 3;          // B rows 0..63

    v8f accg[2][2], acch[2][2];
#pragma unroll
    for (int i = 0; i < 2; ++i)
#pragma unroll
        for (int j = 0; j < 2; ++j) { accg[i][j] = vzero8(); acch[i][j] = vzero8(); }

    // --- stage(buf, k0): 2x A + 1x Bg + 1x Bh async b128 per thread ---
    #define STAGE1(buf, k0)                                                          \
    do {                                                                             \
        ASYNC_LOAD_B128(lds_off(&lA[buf][arow0 * LDA]) + ac0 * 16,                   \
                        Xb + (size_t)(m0 + arow0) * D_MODEL + (k0) + ac0 * 8);       \
        ASYNC_LOAD_B128(lds_off(&lA[buf][arow1 * LDA]) + ac1 * 16,                   \
                        Xb + (size_t)(m0 + arow1) * D_MODEL + (k0) + ac1 * 8);       \
        ASYNC_LOAD_B128(lds_off(&lBg[buf][brow * LDB]) + bc * 16,                    \
                        WinT + (size_t)(n0 + brow) * D_MODEL + (k0) + bc * 8);       \
        ASYNC_LOAD_B128(lds_off(&lBh[buf][brow * LDB]) + bc * 16,                    \
                        WinT + (size_t)(D_MODEL + n0 + brow) * D_MODEL + (k0) + bc * 8); \
    } while (0)

    STAGE1(0, 0);
    for (int i = 0; i < NK; ++i) {
        const int cur = i & 1;
        if (i + 1 < NK) {
            STAGE1(cur ^ 1, (i + 1) * BK);
            WAIT_ASYNC(0x4);     // older stage (4 ops) retired; next stage in flight
        } else {
            WAIT_ASYNC(0x0);
        }
        __syncthreads();

        v16bf af[2], bg[2], bh[2];
#pragma unroll
        for (int mi = 0; mi < 2; ++mi)
            af[mi] = frag_a(&lA[cur][(wm + mi * 16 + id16) * LDA], half);
#pragma unroll
        for (int ni = 0; ni < 2; ++ni) {
            bg[ni] = frag_b(&lBg[cur][(wn + ni * 16 + id16) * LDB], half);
            bh[ni] = frag_b(&lBh[cur][(wn + ni * 16 + id16) * LDB], half);
        }
#pragma unroll
        for (int mi = 0; mi < 2; ++mi)
#pragma unroll
            for (int ni = 0; ni < 2; ++ni) {
                accg[mi][ni] = __builtin_amdgcn_wmma_f32_16x16x32_bf16(
                    false, af[mi], false, bg[ni], (short)0, accg[mi][ni], false, false);
                acch[mi][ni] = __builtin_amdgcn_wmma_f32_16x16x32_bf16(
                    false, af[mi], false, bh[ni], (short)0, acch[mi][ni], false, false);
            }
        __syncthreads();   // everyone done reading buf[cur] before it is restaged
    }
    #undef STAGE1

    // Epilogue: bias + sigmoid(gate) * silu(hidden).
#pragma unroll
    for (int mi = 0; mi < 2; ++mi)
#pragma unroll
        for (int ni = 0; ni < 2; ++ni) {
            int col = n0 + wn + ni * 16 + id16;
            float bgv = bin[col];
            float bhv = bin[D_MODEL + col];
#pragma unroll
            for (int r = 0; r < 8; ++r) {
                int row = m0 + wm + mi * 16 + half * 8 + r;
                float xg = accg[mi][ni][r] + bgv;
                float xh = acch[mi][ni][r] + bhv;
                float sg = 1.0f / (1.0f + __expf(-xg));
                float sh = 1.0f / (1.0f + __expf(-xh));
                G[(size_t)row * D_MODEL + col] = sg * (xh * sh);
            }
        }
}

// ---------------------------------------------------------------------------
// Kernel 2: EMA scan h_t = 0.9 h_{t-1} + 0.1 g_t, chunk-parallel with 256-step
// warm-up (0.9^256 ~ 2e-12, below fp32 noise).  Output bf16 for GEMM2.
// ---------------------------------------------------------------------------
__global__ __launch_bounds__(256)
void mamba_ema_scan(const float* __restrict__ G, bf16_t* __restrict__ H) {
    const int t     = blockIdx.x * 256 + threadIdx.x;  // 0..65535
    const int d     = t & (D_MODEL - 1);
    const int rest  = t >> 10;
    const int b     = rest & (BATCH - 1);
    const int chunk = rest >> 2;                       // 0..15
    const int L     = 256;
    const int s0    = chunk * L;
    const int sw    = (chunk == 0) ? 0 : (s0 - 256);

    const size_t base = (size_t)b * SEQ * D_MODEL + d;
    float h = 0.0f;
    for (int s = sw; s < s0; ++s)
        h = 0.9f * h + 0.1f * G[base + (size_t)s * D_MODEL];
    for (int s = s0; s < s0 + L; ++s) {
        h = 0.9f * h + 0.1f * G[base + (size_t)s * D_MODEL];
        H[base + (size_t)s * D_MODEL] = (bf16_t)h;
    }
}

// ---------------------------------------------------------------------------
// Kernel 3: out = hidden @ W_out + b_out (same async double-buffered scheme)
// ---------------------------------------------------------------------------
__global__ __launch_bounds__(256)
void mamba_gemm2(const bf16_t* __restrict__ H,      // [M,1024] bf16
                 const bf16_t* __restrict__ WoutT,  // [1024,1024] bf16 (n-major)
                 const float* __restrict__ bout,    // [1024]
                 float* __restrict__ Out)           // [M,1024] fp32
{
    __shared__ __align__(16) bf16_t lA[2][BM * LDA];
    __shared__ __align__(16) bf16_t lB[2][BN * LDB];

    const int tid  = threadIdx.x;
    const int m0   = blockIdx.x * BM;
    const int n0   = blockIdx.y * BN;
    const int lane = tid & 31;
    const int w    = tid >> 5;
    const int wm   = (w & 3) * 32;
    const int wn   = (w >> 2) * 32;
    const int half = lane >> 4;
    const int id16 = lane & 15;

    const int arow0 = tid >> 2,         ac0 = tid & 3;
    const int arow1 = (tid + 256) >> 2, ac1 = (tid + 256) & 3;
    const int brow  = tid >> 2,         bc  = tid & 3;

    v8f acc[2][2];
#pragma unroll
    for (int i = 0; i < 2; ++i)
#pragma unroll
        for (int j = 0; j < 2; ++j) acc[i][j] = vzero8();

    #define STAGE2(buf, k0)                                                       \
    do {                                                                          \
        ASYNC_LOAD_B128(lds_off(&lA[buf][arow0 * LDA]) + ac0 * 16,                \
                        H + (size_t)(m0 + arow0) * D_MODEL + (k0) + ac0 * 8);     \
        ASYNC_LOAD_B128(lds_off(&lA[buf][arow1 * LDA]) + ac1 * 16,                \
                        H + (size_t)(m0 + arow1) * D_MODEL + (k0) + ac1 * 8);     \
        ASYNC_LOAD_B128(lds_off(&lB[buf][brow * LDB]) + bc * 16,                  \
                        WoutT + (size_t)(n0 + brow) * D_MODEL + (k0) + bc * 8);   \
    } while (0)

    STAGE2(0, 0);
    for (int i = 0; i < NK; ++i) {
        const int cur = i & 1;
        if (i + 1 < NK) {
            STAGE2(cur ^ 1, (i + 1) * BK);
            WAIT_ASYNC(0x3);
        } else {
            WAIT_ASYNC(0x0);
        }
        __syncthreads();

        v16bf af[2], bf[2];
#pragma unroll
        for (int mi = 0; mi < 2; ++mi)
            af[mi] = frag_a(&lA[cur][(wm + mi * 16 + id16) * LDA], half);
#pragma unroll
        for (int ni = 0; ni < 2; ++ni)
            bf[ni] = frag_b(&lB[cur][(wn + ni * 16 + id16) * LDB], half);
#pragma unroll
        for (int mi = 0; mi < 2; ++mi)
#pragma unroll
            for (int ni = 0; ni < 2; ++ni)
                acc[mi][ni] = __builtin_amdgcn_wmma_f32_16x16x32_bf16(
                    false, af[mi], false, bf[ni], (short)0, acc[mi][ni], false, false);
        __syncthreads();
    }
    #undef STAGE2

#pragma unroll
    for (int mi = 0; mi < 2; ++mi)
#pragma unroll
        for (int ni = 0; ni < 2; ++ni) {
            int col = n0 + wn + ni * 16 + id16;
            float bv = bout[col];
#pragma unroll
            for (int r = 0; r < 8; ++r) {
                int row = m0 + wm + mi * 16 + half * 8 + r;
                Out[(size_t)row * D_MODEL + col] = acc[mi][ni][r] + bv;
            }
        }
}

// ---------------------------------------------------------------------------
extern "C" void kernel_launch(void* const* d_in, const int* in_sizes, int n_in,
                              void* d_out, int out_size, void* d_ws, size_t ws_size,
                              hipStream_t stream) {
    (void)in_sizes; (void)n_in; (void)out_size; (void)ws_size;
    const float* x    = (const float*)d_in[0];  // [4,4096,1024]
    const float* Win  = (const float*)d_in[1];  // [1024,2048]
    const float* bin  = (const float*)d_in[2];  // [2048]
    const float* Wout = (const float*)d_in[3];  // [1024,1024]
    const float* bout = (const float*)d_in[4];  // [1024]

    float* out = (float*)d_out;
    float* G   = out;   // d_out doubles as gated-activation scratch (consumed by scan)

    char* ws = (char*)d_ws;
    bf16_t* Xb    = (bf16_t*)(ws);                          // 32 MB
    bf16_t* Hh    = (bf16_t*)(ws + (size_t)32 * 1024 * 1024); // 32 MB
    bf16_t* WinT  = (bf16_t*)(ws + (size_t)64 * 1024 * 1024); //  4 MB  [2048][1024]
    bf16_t* WoutT = (bf16_t*)(ws + (size_t)68 * 1024 * 1024); //  2 MB  [1024][1024]

    dim3 blk(256, 1, 1);

    // One-time precision/layout prep (bf16, weights transposed to n-major).
    cvt_f32_to_bf16<<<dim3((M_TOTAL * D_MODEL) / (256 * 8), 1, 1), blk, 0, stream>>>(x, Xb);
    transpose_cvt  <<<dim3(TWO_D / 32, D_MODEL / 32, 1), blk, 0, stream>>>(Win, WinT, D_MODEL, TWO_D);
    transpose_cvt  <<<dim3(D_MODEL / 32, D_MODEL / 32, 1), blk, 0, stream>>>(Wout, WoutT, D_MODEL, D_MODEL);

    dim3 grid_gemm(M_TOTAL / BM, D_MODEL / BN, 1);   // 128 x 16
    mamba_gemm1_gated<<<grid_gemm, blk, 0, stream>>>(Xb, WinT, bin, G);
    mamba_ema_scan  <<<dim3(256, 1, 1), blk, 0, stream>>>(G, Hh);
    mamba_gemm2     <<<grid_gemm, blk, 0, stream>>>(Hh, WoutT, bout, out);
}
